// TFGWLayer_74964359184849
// MI455X (gfx1250) — compile-verified, matrix operand order checked by reading, not run on hardware
//
#include <hip/hip_runtime.h>

typedef __attribute__((ext_vector_type(16))) _Float16 v16h;
typedef __attribute__((ext_vector_type(8)))  float    v8f;
typedef __attribute__((ext_vector_type(4)))  unsigned v4u;
typedef __attribute__((ext_vector_type(4)))  int      v4i;

#define NN    6144      // nodes
#define NE    98304     // edges
#define NF    128       // features
#define TP    16        // templates
#define MN    10        // template nodes
#define NCG   5
#define NSINK 50
#define WPR   (NN/32)   // 192 bitmask words per row
#define NCOL  (TP*MN)   // 160
#define TNM   (TP*NN*MN) // 983040
#define BTILE (2*4*32*16) // halfs per B wave-tile (hi+lo, 4 k-steps) = 4096 (8KB)

#if __has_builtin(__builtin_amdgcn_global_load_async_to_lds_b128) && \
    __has_builtin(__builtin_amdgcn_s_wait_asynccnt)
#define TFGW_ASYNC 1
#define GAS1 __attribute__((address_space(1)))
#define LAS3 __attribute__((address_space(3)))
#endif

// ---------------- adjacency construction (dedup via bitset -> CSR) -------------

__global__ void k_clear_u32(unsigned* p, int n) {
  int i = blockIdx.x*blockDim.x + threadIdx.x;
  if (i < n) p[i] = 0u;
}

__global__ void k_setbits(const long long* __restrict__ ei, unsigned* __restrict__ bits) {
  int e = blockIdx.x*blockDim.x + threadIdx.x;
  if (e >= NE) return;
  int i = (int)ei[e];
  int j = (int)ei[NE + e];
  atomicOr(&bits[(size_t)i*WPR + (j >> 5)], 1u << (j & 31));
  atomicOr(&bits[(size_t)j*WPR + (i >> 5)], 1u << (i & 31));
}

__global__ void k_degcsum(const unsigned* __restrict__ bits, unsigned* __restrict__ deg,
                          float* __restrict__ csum) {
  int i = blockIdx.x*blockDim.x + threadIdx.x;
  if (i >= NN) return;
  unsigned d = 0;
  for (int w = 0; w < WPR; ++w) d += (unsigned)__popc(bits[(size_t)i*WPR + w]);
  deg[i] = d;
  csum[i] = (float)d * (1.0f / (float)NN);   // (C1^2 @ p)[i], C1 binary
}

__global__ __launch_bounds__(1024) void k_scan(const unsigned* __restrict__ deg,
                                               unsigned* __restrict__ offs) {
  const int t = threadIdx.x;              // 1024 threads, 6 rows each
  __shared__ unsigned sP[1024];
  unsigned loc[6];
  unsigned s = 0;
  #pragma unroll
  for (int k = 0; k < 6; ++k) { loc[k] = deg[t*6 + k]; s += loc[k]; }
  sP[t] = s; __syncthreads();
  for (int st = 1; st < 1024; st <<= 1) {
    unsigned v = 0;
    if (t >= st) v = sP[t - st];
    __syncthreads();
    sP[t] += v;
    __syncthreads();
  }
  unsigned run = sP[t] - s;               // exclusive prefix of this chunk
  #pragma unroll
  for (int k = 0; k < 6; ++k) { offs[t*6 + k] = run; run += loc[k]; }
  if (t == 1023) offs[NN] = sP[1023];
}

__global__ void k_fillcols(const unsigned* __restrict__ bits, const unsigned* __restrict__ offs,
                           unsigned* __restrict__ cols) {
  int i = blockIdx.x*blockDim.x + threadIdx.x;
  if (i >= NN) return;
  unsigned pos = offs[i];
  for (int w = 0; w < WPR; ++w) {
    unsigned b = bits[(size_t)i*WPR + w];
    while (b) {
      int bit = __ffs(b) - 1;
      cols[pos++] = (unsigned)(w*32 + bit);
      b &= b - 1;
    }
  }
}

// ---------------- small scalar precompute -------------------------------------

__global__ void k_scalars(const float* __restrict__ tpl, const float* __restrict__ tf,
                          const float* __restrict__ q0, const float* __restrict__ a0,
                          float* alphaP, float* q, float* logq, float* c2q, float* tfsq) {
  const int tid = threadIdx.x;
  if (tid == 0) alphaP[0] = 1.0f / (1.0f + __expf(-a0[0]));
  if (tid < NCOL) {
    const int t = tid / MN;
    float m = -3.0e38f;
    for (int k = 0; k < MN; ++k) m = fmaxf(m, q0[t*MN + k]);
    float s = 0.f;
    for (int k = 0; k < MN; ++k) s += __expf(q0[t*MN + k] - m);
    q[tid]    = __expf(q0[tid] - m) / s;
    logq[tid] = (q0[tid] - m) - __logf(s);
    float ts = 0.f;
    for (int k = 0; k < NF; ++k) { float v = tf[(size_t)tid*NF + k]; ts += v*v; }
    tfsq[tid] = ts;
  }
  __syncthreads();
  if (tid < NCOL) {
    const int t = tid / MN, j = tid % MN;
    float s = 0.f;
    for (int k = 0; k < MN; ++k) { float c = tpl[t*100 + j*10 + k]; s += c*c*q[t*MN + k]; }
    c2q[tid] = s;                         // (C2^2 @ q)[t][j]
  }
}

__global__ void k_xsq(const float* __restrict__ x, float* __restrict__ xsq) {
  int i = blockIdx.x*blockDim.x + threadIdx.x;
  if (i >= NN) return;
  float s = 0.f;
  for (int k = 0; k < NF; ++k) { float v = x[(size_t)i*NF + k]; s += v*v; }
  xsq[i] = s;
}

// split f32 -> f16 hi + f16 residual (≈22 mantissa bits recovered with 3 WMMAs)
__global__ void k_cvt_x(const float* __restrict__ x, _Float16* __restrict__ xh,
                        _Float16* __restrict__ xl) {
  int i = blockIdx.x*blockDim.x + threadIdx.x;
  if (i >= NN*NF) return;
  float v = x[i];
  _Float16 h = (_Float16)v;
  xh[i] = h;
  xl[i] = (_Float16)(v - (float)h);
}

// B = TF^T, split-f16, pre-swizzled into WMMA fragment order so the GEMM can
// move it with contiguous 128-bit (and async-to-LDS) transfers:
//   bfrag[tn][h][kb][lane][16 halfs],  h=0:hi, 1:lo residual
__global__ void k_bfrag(const float* __restrict__ tf, _Float16* __restrict__ bfrag) {
  int idx = blockIdx.x*blockDim.x + threadIdx.x;   // 10*2*4*32*8 = 20480 half-pairs
  if (idx >= 10*2*4*32*8) return;
  const int v    = idx & 7;
  const int lane = (idx >> 3) & 31;
  const int kb   = (idx >> 8) & 3;
  const int h    = (idx >> 10) & 1;
  const int tn   = idx >> 11;
  const int lm = lane & 15, lh4 = lane >> 4;
  const int col = tn*16 + lm;                               // global column (t*MN+j)
  const int kk  = kb*32 + 2*(v & 3) + 16*(v >> 2) + 8*lh4;  // ISA 16-bit B K-pattern
  float v0 = tf[(size_t)col*NF + kk];
  float v1 = tf[(size_t)col*NF + kk + 1];
  _Float16 h0 = (_Float16)v0, h1 = (_Float16)v1;
  _Float16 e0 = (h == 0) ? h0 : (_Float16)(v0 - (float)h0);
  _Float16 e1 = (h == 0) ? h1 : (_Float16)(v1 - (float)h1);
  size_t base = ((((size_t)tn*2 + h)*4 + kb)*32 + lane)*16 + 2*v;
  bfrag[base]   = e0;
  bfrag[base+1] = e1;
}

// ---------------- WMMA GEMM: XTF = x @ TF^T  (6144x128 @ 128x160) -------------
// one 16x16 tile per wave32; split-f16: AhBh + AhBl + AlBh.
// A fragments: 2 x b128 per fragment straight from global (row-major K pairs are
// contiguous).  B fragments: wave's 8KB tile staged into LDS via async copies
// (ASYNCcnt-tracked) when the gfx1250 async builtins are available, else direct
// b128 global loads.

__global__ __launch_bounds__(256) void k_gemm(const _Float16* __restrict__ xh,
                                              const _Float16* __restrict__ xl,
                                              const _Float16* __restrict__ bfrag,
                                              float* __restrict__ out) {
  const int wv   = threadIdx.x >> 5;
  const int wave = blockIdx.x*(blockDim.x >> 5) + wv;
  const int lane = threadIdx.x & 31;
  const int lm = lane & 15, lh = lane >> 4;
#ifdef TFGW_ASYNC
  __shared__ _Float16 sB[8][BTILE];        // 8 waves * 8KB
#endif
  if (wave >= (NN/16)*(NCOL/16)) return;   // wave-uniform guard (never taken: exact grid)
  const int tm = wave / (NCOL/16);
  const int tn = wave % (NCOL/16);
  const int row = tm*16 + lm;              // A row (M) for this lane
  const int col = tn*16 + lm;              // output column for this lane
  const _Float16* btile = bfrag + (size_t)tn*BTILE;
#ifdef TFGW_ASYNC
  // stage this wave's B tile: 512 b128 units, 16 per lane, ASYNCcnt-tracked
  #pragma unroll
  for (int c = 0; c < 16; ++c) {
    const int u = c*32 + lane;
    __builtin_amdgcn_global_load_async_to_lds_b128(
        (GAS1 v4i*)(btile + (size_t)u*8),
        (LAS3 v4i*)(&sB[wv][u*8]),
        0, 0);
  }
  __builtin_amdgcn_s_wait_asynccnt(0);
  const _Float16* bbase = &sB[wv][0];
#else
  const _Float16* bbase = btile;
#endif
  typedef union { v16h h; v4u u[2]; } fragA;
  v8f acc = {};
  #pragma unroll
  for (int kb32 = 0; kb32 < 4; ++kb32) {
    const int kb = kb32*32;
    fragA aH, aL;
    // A 16-bit 16x32 layout: lane's 16 halfs = two contiguous 16B runs
    aH.u[0] = *(const v4u*)(xh + (size_t)row*NF + kb + 8*lh);
    aH.u[1] = *(const v4u*)(xh + (size_t)row*NF + kb + 16 + 8*lh);
    aL.u[0] = *(const v4u*)(xl + (size_t)row*NF + kb + 8*lh);
    aL.u[1] = *(const v4u*)(xl + (size_t)row*NF + kb + 16 + 8*lh);
    const v16h bH = *(const v16h*)(bbase + ((0*4 + kb32)*32 + lane)*16);
    const v16h bL = *(const v16h*)(bbase + ((1*4 + kb32)*32 + lane)*16);
    acc = __builtin_amdgcn_wmma_f32_16x16x32_f16(false, aH.h, false, bH, (short)0, acc, false, false);
    acc = __builtin_amdgcn_wmma_f32_16x16x32_f16(false, aH.h, false, bL, (short)0, acc, false, false);
    acc = __builtin_amdgcn_wmma_f32_16x16x32_f16(false, aL.h, false, bH, (short)0, acc, false, false);
  }
  // C/D layout: VGPR r -> M = r + 8*(lane>>4), N = lane&15
  #pragma unroll
  for (int r = 0; r < 8; ++r)
    out[(size_t)(tm*16 + r + 8*lh)*NCOL + col] = acc[r];
}

// ---------------- elementwise setup kernels -----------------------------------

__global__ void k_M(const float* __restrict__ xsq, const float* __restrict__ tfsq,
                    const float* __restrict__ xtf, float* __restrict__ Mm) {
  int idx = blockIdx.x*blockDim.x + threadIdx.x;
  if (idx >= TNM) return;
  int t = idx / (NN*MN);
  int r = idx % (NN*MN);
  int i = r / MN, j = r % MN;
  int c = t*MN + j;
  Mm[idx] = xsq[i] + tfsq[c] - 2.0f * xtf[(size_t)i*NCOL + c];
}

__global__ void k_initT(const float* __restrict__ q, float* __restrict__ T) {
  int idx = blockIdx.x*blockDim.x + threadIdx.x;
  if (idx >= TNM) return;
  int t = idx / (NN*MN);
  int j = idx % MN;
  T[idx] = (1.0f / (float)NN) * q[t*MN + j];
}

// ---------------- structure product:  OUT = C1 @ X @ C2  (X = X1 or X1-X2) ----

__global__ __launch_bounds__(128) void k_spmm(const float* __restrict__ X1,
                                              const float* __restrict__ X2,
                                              const float* __restrict__ tpl,
                                              const unsigned* __restrict__ offs,
                                              const unsigned* __restrict__ cols,
                                              float* __restrict__ OUT) {
  const int t = blockIdx.y;
  const int i = blockIdx.x*blockDim.x + threadIdx.x;
  __shared__ float sC2[100];
  if (threadIdx.x < 100) sC2[threadIdx.x] = tpl[t*100 + threadIdx.x];
  __syncthreads();
  float acc[MN];
  #pragma unroll
  for (int c = 0; c < MN; ++c) acc[c] = 0.f;
  const unsigned e0 = offs[i], e1 = offs[i+1];
  const size_t tb = (size_t)t*NN*MN;
  if (X2) {
    for (unsigned e = e0; e < e1; ++e) {
      const size_t rb = tb + (size_t)cols[e]*MN;
      #pragma unroll
      for (int c = 0; c < MN; ++c) acc[c] += X1[rb+c] - X2[rb+c];
    }
  } else {
    for (unsigned e = e0; e < e1; ++e) {
      const size_t rb = tb + (size_t)cols[e]*MN;
      #pragma unroll
      for (int c = 0; c < MN; ++c) acc[c] += X1[rb+c];
    }
  }
  #pragma unroll
  for (int c = 0; c < MN; ++c) {
    float o = 0.f;
    #pragma unroll
    for (int k = 0; k < MN; ++k) o += acc[k]*sC2[k*10 + c];
    OUT[tb + (size_t)i*MN + c] = o;
  }
}

// ---------------- gradient + per-block max|grad| ------------------------------

__global__ __launch_bounds__(512) void k_grad(const float* __restrict__ Mm,
                                              const float* __restrict__ csum,
                                              const float* __restrict__ c2q,
                                              const float* __restrict__ CTC,
                                              const float* __restrict__ alphaP,
                                              float* __restrict__ grad,
                                              float* __restrict__ gmaxp) {
  const int t = blockIdx.y;
  const int flat = blockIdx.x*512 + threadIdx.x;       // < 61440
  const int i = flat / MN, j = flat % MN;
  const float a = alphaP[0];
  const size_t idx = (size_t)t*NN*MN + flat;
  float g = (1.f - a)*Mm[idx] + 2.f*a*(csum[i] + c2q[t*MN + j]) - 4.f*a*CTC[idx];
  grad[idx] = g;
  __shared__ float sB[512];
  sB[threadIdx.x] = fabsf(g); __syncthreads();
  for (int s = 256; s > 0; s >>= 1) {
    if (threadIdx.x < s) sB[threadIdx.x] = fmaxf(sB[threadIdx.x], sB[threadIdx.x + s]);
    __syncthreads();
  }
  if (threadIdx.x == 0) gmaxp[t*gridDim.x + blockIdx.x] = sB[0];
}

// ---------------- log-domain Sinkhorn: one workgroup per template -------------
// 512 threads * 12 rows; f in registers, g & lse partials through LDS.

__global__ __launch_bounds__(512) void k_sinkhorn(const float* __restrict__ grad,
                                                  const float* __restrict__ gmaxp,
                                                  const float* __restrict__ logq,
                                                  float* __restrict__ Tn) {
  const int t = blockIdx.x, tid = threadIdx.x;
  __shared__ float sM[512*MN];
  __shared__ float sS[512*MN];
  __shared__ float sG[MN];
  __shared__ float sInvReg;
  // finalize reg = 0.05*max|grad| + 1e-8 from per-block partials
  float mx = 0.f;
  for (int b = tid; b < 120; b += 512) mx = fmaxf(mx, gmaxp[t*120 + b]);
  sM[tid] = mx; __syncthreads();
  for (int s = 256; s > 0; s >>= 1) {
    if (tid < s) sM[tid] = fmaxf(sM[tid], sM[tid + s]);
    __syncthreads();
  }
  if (tid == 0) sInvReg = 1.0f / (0.05f*sM[0] + 1e-8f);
  __syncthreads();
  const float invReg = sInvReg;
  const float logp = -__logf((float)NN);
  float lq[MN];
  #pragma unroll
  for (int j = 0; j < MN; ++j) lq[j] = logq[t*MN + j];
  const float* K = grad + (size_t)t*NN*MN;             // K_val = -K[i,j]*invReg
  float f[12];
  #pragma unroll
  for (int r = 0; r < 12; ++r) f[r] = 0.f;
  float g[MN];
  for (int it = 0; it <= NSINK; ++it) {
    // ---- g[j] = lq[j] - lse_i( Kv + f ) : streaming lse then LDS tree reduce
    float m[MN], s[MN];
    #pragma unroll
    for (int j = 0; j < MN; ++j) { m[j] = -3.0e38f; s[j] = 0.f; }
    for (int r = 0; r < 12; ++r) {
      const int i = tid + r*512;
      const float fr = f[r];
      #pragma unroll
      for (int j = 0; j < MN; ++j) {
        float v = -K[(size_t)i*MN + j]*invReg + fr;
        float M2 = fmaxf(m[j], v);
        s[j] = s[j]*__expf(m[j] - M2) + __expf(v - M2);
        m[j] = M2;
      }
    }
    #pragma unroll
    for (int j = 0; j < MN; ++j) { sM[tid*MN + j] = m[j]; sS[tid*MN + j] = s[j]; }
    __syncthreads();
    for (int st = 256; st > 0; st >>= 1) {
      if (tid < st) {
        #pragma unroll
        for (int j = 0; j < MN; ++j) {
          float m1 = sM[tid*MN + j],      s1 = sS[tid*MN + j];
          float m2 = sM[(tid+st)*MN + j], s2 = sS[(tid+st)*MN + j];
          float M2 = fmaxf(m1, m2);
          sS[tid*MN + j] = s1*__expf(m1 - M2) + s2*__expf(m2 - M2);
          sM[tid*MN + j] = M2;
        }
      }
      __syncthreads();
    }
    if (tid < MN) sG[tid] = lq[tid] - (__logf(sS[tid]) + sM[tid]);
    __syncthreads();
    #pragma unroll
    for (int j = 0; j < MN; ++j) g[j] = sG[j];
    __syncthreads();
    if (it == NSINK) break;
    // ---- f[i] = logp - lse_j( Kv + g ) : local 10-wide lse
    for (int r = 0; r < 12; ++r) {
      const int i = tid + r*512;
      float vv[MN], mm = -3.0e38f;
      #pragma unroll
      for (int j = 0; j < MN; ++j) {
        vv[j] = -K[(size_t)i*MN + j]*invReg + g[j];
        mm = fmaxf(mm, vv[j]);
      }
      float ss = 0.f;
      #pragma unroll
      for (int j = 0; j < MN; ++j) ss += __expf(vv[j] - mm);
      f[r] = logp - (__logf(ss) + mm);
    }
  }
  // Tn = exp(f + Kv + g)
  for (int r = 0; r < 12; ++r) {
    const int i = tid + r*512;
    #pragma unroll
    for (int j = 0; j < MN; ++j) {
      float Kv = -K[(size_t)i*MN + j]*invReg;
      Tn[(size_t)t*NN*MN + (size_t)i*MN + j] = __expf(f[r] + Kv + g[j]);
    }
  }
}

// ---------------- line-search dot products ------------------------------------

__global__ __launch_bounds__(512) void k_dots(const float* __restrict__ Tn,
                                              const float* __restrict__ T,
                                              const float* __restrict__ Mm,
                                              const float* __restrict__ csum,
                                              const float* __restrict__ c2q,
                                              const float* __restrict__ CTC,
                                              const float* __restrict__ CdTC,
                                              float* __restrict__ dotp) {
  const int t = blockIdx.y;
  const int flat = blockIdx.x*512 + threadIdx.x;
  const int i = flat / MN, j = flat % MN;
  const size_t idx = (size_t)t*NN*MN + flat;
  const float dT = Tn[idx] - T[idx];
  float v[4];
  v[0] = CdTC[idx]*dT;                     // sum C1dTC2 . dT
  v[1] = Mm[idx]*dT;                       // sum M . dT
  v[2] = (csum[i] + c2q[t*MN + j])*dT;     // sum constC . dT
  v[3] = CTC[idx]*dT;                      // sum C1TC2 . dT
  __shared__ float sB[512];
  #pragma unroll
  for (int k = 0; k < 4; ++k) {
    sB[threadIdx.x] = v[k]; __syncthreads();
    for (int s = 256; s > 0; s >>= 1) {
      if (threadIdx.x < s) sB[threadIdx.x] += sB[threadIdx.x + s];
      __syncthreads();
    }
    if (threadIdx.x == 0) dotp[((size_t)t*120 + blockIdx.x)*4 + k] = sB[0];
    __syncthreads();
  }
}

__global__ __launch_bounds__(128) void k_tau(const float* __restrict__ dotp,
                                             const float* __restrict__ alphaP,
                                             float* __restrict__ tau) {
  const int t = blockIdx.x, tid = threadIdx.x;
  float s[4] = {0.f, 0.f, 0.f, 0.f};
  for (int b = tid; b < 120; b += 128) {
    #pragma unroll
    for (int k = 0; k < 4; ++k) s[k] += dotp[((size_t)t*120 + b)*4 + k];
  }
  __shared__ float sB[128*4];
  #pragma unroll
  for (int k = 0; k < 4; ++k) sB[k*128 + tid] = s[k];
  __syncthreads();
  for (int st = 64; st > 0; st >>= 1) {
    if (tid < st) {
      #pragma unroll
      for (int k = 0; k < 4; ++k) sB[k*128 + tid] += sB[k*128 + tid + st];
    }
    __syncthreads();
  }
  if (tid == 0) {
    const float al = alphaP[0];
    float a = -2.f*al*sB[0];
    float b = (1.f - al)*sB[128] + al*(sB[256] - 4.f*sB[384]);
    float tv;
    if (a > 0.f) { tv = -b/(2.f*a + 1e-16f); tv = fminf(fmaxf(tv, 0.f), 1.f); }
    else tv = (a + b < 0.f) ? 1.f : 0.f;
    tau[t] = tv;
  }
}

__global__ void k_update(float* __restrict__ T, const float* __restrict__ Tn,
                         const float* __restrict__ tau) {
  int idx = blockIdx.x*blockDim.x + threadIdx.x;
  if (idx >= TNM) return;
  int t = idx / (NN*MN);
  float Tv = T[idx];
  T[idx] = Tv + tau[t]*(Tn[idx] - Tv);
}

// ---------------- final objective ---------------------------------------------

__global__ __launch_bounds__(512) void k_fdots(const float* __restrict__ T,
                                               const float* __restrict__ Mm,
                                               const float* __restrict__ csum,
                                               const float* __restrict__ c2q,
                                               const float* __restrict__ CTC,
                                               float* __restrict__ dotp) {
  const int t = blockIdx.y;
  const int flat = blockIdx.x*512 + threadIdx.x;
  const int i = flat / MN, j = flat % MN;
  const size_t idx = (size_t)t*NN*MN + flat;
  const float Tv = T[idx];
  float v[3];
  v[0] = Mm[idx]*Tv;                       // wass
  v[1] = (csum[i] + c2q[t*MN + j])*Tv;     // sum constC . T
  v[2] = CTC[idx]*Tv;                      // sum C1TC2 . T
  __shared__ float sB[512];
  #pragma unroll
  for (int k = 0; k < 3; ++k) {
    sB[threadIdx.x] = v[k]; __syncthreads();
    for (int s = 256; s > 0; s >>= 1) {
      if (threadIdx.x < s) sB[threadIdx.x] += sB[threadIdx.x + s];
      __syncthreads();
    }
    if (threadIdx.x == 0) dotp[((size_t)t*120 + blockIdx.x)*3 + k] = sB[0];
    __syncthreads();
  }
}

__global__ __launch_bounds__(128) void k_out(const float* __restrict__ dotp,
                                             const float* __restrict__ alphaP,
                                             float* __restrict__ out) {
  const int t = blockIdx.x, tid = threadIdx.x;
  float s[3] = {0.f, 0.f, 0.f};
  for (int b = tid; b < 120; b += 128) {
    #pragma unroll
    for (int k = 0; k < 3; ++k) s[k] += dotp[((size_t)t*120 + b)*3 + k];
  }
  __shared__ float sB[128*3];
  #pragma unroll
  for (int k = 0; k < 3; ++k) sB[k*128 + tid] = s[k];
  __syncthreads();
  for (int st = 64; st > 0; st >>= 1) {
    if (tid < st) {
      #pragma unroll
      for (int k = 0; k < 3; ++k) sB[k*128 + tid] += sB[k*128 + tid + st];
    }
    __syncthreads();
  }
  if (tid == 0) {
    const float al = alphaP[0];
    float wass = sB[0];
    float gw   = sB[128] - 2.f*sB[256];
    out[t] = (1.f - al)*wass + al*gw;
  }
}

// ---------------- host orchestration ------------------------------------------

extern "C" void kernel_launch(void* const* d_in, const int* in_sizes, int n_in,
                              void* d_out, int out_size, void* d_ws, size_t ws_size,
                              hipStream_t stream) {
  const float*     x   = (const float*)d_in[0];
  const long long* ei  = (const long long*)d_in[1];   // int64 [2, NE]
  const float*     tpl = (const float*)d_in[2];       // [TP, MN, MN]
  const float*     tf  = (const float*)d_in[3];       // [TP, MN, NF]
  const float*     q0  = (const float*)d_in[4];       // [TP, MN]
  const float*     a0  = (const float*)d_in[5];       // [1]
  float* out = (float*)d_out;

  char* base = (char*)d_ws;
  size_t off = 0;
  auto alloc = [&](size_t bytes) -> char* {
    char* p = base + off;
    off = (off + bytes + 255) & ~(size_t)255;
    return p;
  };
  unsigned* bits = (unsigned*)alloc((size_t)NN*WPR*4);       // 4.72 MB (reused as grad)
  unsigned* deg  = (unsigned*)alloc((size_t)NN*4);
  float*    csum = (float*)   alloc((size_t)NN*4);
  unsigned* offs = (unsigned*)alloc((size_t)(NN+1)*4);
  unsigned* cols = (unsigned*)alloc((size_t)2*NE*4);
  _Float16* xh   = (_Float16*)alloc((size_t)NN*NF*2);
  _Float16* xl   = (_Float16*)alloc((size_t)NN*NF*2);
  _Float16* bfr  = (_Float16*)alloc((size_t)10*BTILE*2);     // fragment-ordered B
  float*    xtf  = (float*)   alloc((size_t)NN*NCOL*4);      // 3.93 MB (reused as C1TC2)
  float*    xsq  = (float*)   alloc((size_t)NN*4);
  float*    tfsq = (float*)   alloc((size_t)NCOL*4);
  float*    alphaP=(float*)   alloc(256);
  float*    q    = (float*)   alloc((size_t)NCOL*4);
  float*    logq = (float*)   alloc((size_t)NCOL*4);
  float*    c2q  = (float*)   alloc((size_t)NCOL*4);
  float*    Mm   = (float*)   alloc((size_t)TNM*4);
  float*    T    = (float*)   alloc((size_t)TNM*4);
  float*    Tn   = (float*)   alloc((size_t)TNM*4);
  float*    CdTC = (float*)   alloc((size_t)TNM*4);
  float*    gmaxp= (float*)   alloc((size_t)TP*120*4);
  float*    dotp = (float*)   alloc((size_t)TP*120*4*4);
  float*    tau  = (float*)   alloc((size_t)TP*4);
  float*    CTC  = xtf;              // alias: xtf dead after k_M
  float*    grad = (float*)bits;     // alias: bits dead after k_fillcols

  // --- build dedup'd symmetric CSR of C1 ---
  k_clear_u32<<<(NN*WPR + 255)/256, 256, 0, stream>>>(bits, NN*WPR);
  k_setbits  <<<(NE + 255)/256,     256, 0, stream>>>(ei, bits);
  k_degcsum  <<<(NN + 255)/256,     256, 0, stream>>>(bits, deg, csum);
  k_scan     <<<1, 1024, 0, stream>>>(deg, offs);
  k_fillcols <<<(NN + 255)/256,     256, 0, stream>>>(bits, offs, cols);

  // --- scalar / feature precompute ---
  k_scalars<<<1, 256, 0, stream>>>(tpl, tf, q0, a0, alphaP, q, logq, c2q, tfsq);
  k_xsq    <<<(NN + 255)/256, 256, 0, stream>>>(x, xsq);
  k_cvt_x  <<<(NN*NF + 255)/256, 256, 0, stream>>>(x, xh, xl);
  k_bfrag  <<<(10*2*4*32*8 + 255)/256, 256, 0, stream>>>(tf, bfr);
  k_gemm   <<<480, 256, 0, stream>>>(xh, xl, bfr, xtf);
  k_M      <<<(TNM + 255)/256, 256, 0, stream>>>(xsq, tfsq, xtf, Mm);
  k_initT  <<<(TNM + 255)/256, 256, 0, stream>>>(q, T);

  const dim3 gSp(NN/128, TP), bSp(128);
  const dim3 gEl(120, TP),    bEl(512);

  // --- conditional-gradient (Frank-Wolfe) loop ---
  for (int it = 0; it < NCG; ++it) {
    k_spmm    <<<gSp, bSp, 0, stream>>>(T, nullptr, tpl, offs, cols, CTC);
    k_grad    <<<gEl, bEl, 0, stream>>>(Mm, csum, c2q, CTC, alphaP, grad, gmaxp);
    k_sinkhorn<<<TP, 512, 0, stream>>>(grad, gmaxp, logq, Tn);
    k_spmm    <<<gSp, bSp, 0, stream>>>(Tn, T, tpl, offs, cols, CdTC);
    k_dots    <<<gEl, bEl, 0, stream>>>(Tn, T, Mm, csum, c2q, CTC, CdTC, dotp);
    k_tau     <<<TP, 128, 0, stream>>>(dotp, alphaP, tau);
    k_update  <<<(TNM + 255)/256, 256, 0, stream>>>(T, Tn, tau);
  }

  // --- final objective ---
  k_spmm <<<gSp, bSp, 0, stream>>>(T, nullptr, tpl, offs, cols, CTC);
  k_fdots<<<gEl, bEl, 0, stream>>>(T, Mm, csum, c2q, CTC, dotp);
  k_out  <<<TP, 128, 0, stream>>>(dotp, alphaP, out);

  (void)in_sizes; (void)n_in; (void)out_size; (void)ws_size;
}
